// GroupedQueryAttention_11948599017915
// MI455X (gfx1250) — compile-verified, matrix operand order checked by reading, not run on hardware
//
#include <hip/hip_runtime.h>

// ---------------- problem constants ----------------
#define DIMN   2048
#define HDN    128
#define HEADS  16
#define KVHN   4
#define SEQN   2048
#define BATCHN 4
#define KVDN   (KVHN * HDN)   // 512
#define EPSF   1e-6f

#define ASG __attribute__((address_space(1)))
#define ASL __attribute__((address_space(3)))

typedef __attribute__((ext_vector_type(16))) __bf16 v16bf;
typedef __attribute__((ext_vector_type(8)))  float  v8f;
typedef __attribute__((ext_vector_type(4)))  unsigned int u32x4;
typedef __attribute__((ext_vector_type(8)))  int i32x8;
typedef __attribute__((ext_vector_type(4)))  int i32x4;
typedef __attribute__((ext_vector_type(4)))  int v4i;

// ------- feature probes (compile-safe: fall back to plain loads) -------
#if defined(__has_builtin)
#  if __has_builtin(__builtin_amdgcn_global_load_async_to_lds_b128)
#    define HAS_ASYNC 1
#  endif
#  if __has_builtin(__builtin_amdgcn_tensor_load_to_lds)
#    define HAS_TDM 1
#  endif
#endif
#ifndef HAS_ASYNC
#define HAS_ASYNC 0
#endif
#ifndef HAS_TDM
#define HAS_TDM 0
#endif

union FragU { v16bf v; uint4 q[2]; };

__device__ __forceinline__ unsigned short f2bf(float f) {
  unsigned u = __float_as_uint(f);
  u += 0x7fffu + ((u >> 16) & 1u);          // round-to-nearest-even
  return (unsigned short)(u >> 16);
}

// 16B global -> LDS async copy (ASYNCcnt tracked), per-lane.
// Probe-derived signature: (int4 AS1*, int4 AS3*, imm offset, imm cpol).
__device__ __forceinline__ void async_cp16(const void* g, void* l) {
#if HAS_ASYNC
  __builtin_amdgcn_global_load_async_to_lds_b128(
      (ASG v4i*)(unsigned long long)g,
      (ASL v4i*)(unsigned)(unsigned long long)l, 0, 0);
#else
  (void)g; (void)l;
#endif
}

// TDM: DMA a 2-D bf16 tile [tile1 rows x tile0 elems] (row stride stride0
// elems in global) into LDS at ldsOff, padding (amtCode+1) DWORDs after every
// 2^(ivCode+1) DWORDs. One instruction per wave; TENSORcnt tracked.
__device__ __forceinline__ void tdm_load_2d(const void* gsrc, unsigned ldsOff,
                                            unsigned tile0, unsigned tile1,
                                            unsigned tensor0, unsigned tensor1,
                                            unsigned stride0,
                                            unsigned ivCode, unsigned amtCode) {
#if HAS_TDM
  unsigned long long ga = (unsigned long long)gsrc;
  u32x4 g0;
  g0[0] = 1u;                                            // count=1 (user D#)
  g0[1] = ldsOff;                                        // lds_addr
  g0[2] = (unsigned)(ga & 0xffffffffu);                  // global_addr lo
  g0[3] = (unsigned)((ga >> 32) & 0x01ffffffu) | (2u << 30);  // hi + type=2
  i32x8 g1;
  g1[0] = (int)((1u << 16) |                             // data_size = 2B
                (1u << 20) |                             // pad_enable
                (ivCode << 22) | (amtCode << 25));       // pad interval/amount
  g1[1] = (int)(tensor0 << 16);                          // tensor_dim0 lo16
  g1[2] = (int)((tensor0 >> 16) | (tensor1 << 16));      // dim0 hi / dim1 lo
  g1[3] = (int)((tensor1 >> 16) | (tile0 << 16));        // dim1 hi / tile_dim0
  g1[4] = (int)(tile1 & 0xffffu);                        // tile_dim1 (dim2=0)
  g1[5] = (int)stride0;                                  // tensor_dim0_stride lo
  g1[6] = 0;
  g1[7] = 0;
  i32x4 z4 = {0, 0, 0, 0};
#if defined(__clang_major__) && (__clang_major__ >= 23)
  i32x8 z8 = {0, 0, 0, 0, 0, 0, 0, 0};
  __builtin_amdgcn_tensor_load_to_lds(g0, g1, z4, z4, z8, 0);
#else
  __builtin_amdgcn_tensor_load_to_lds(g0, g1, z4, z4, 0);
#endif
#else
  (void)gsrc; (void)ldsOff; (void)tile0; (void)tile1;
  (void)tensor0; (void)tensor1; (void)stride0; (void)ivCode; (void)amtCode;
#endif
}

__device__ __forceinline__ unsigned lds_off(const void* p) {
  return (unsigned)(unsigned long long)p;   // low 32 bits = LDS byte address
}

// Load one 16x32 bf16 A/B fragment from LDS (row-major, stride elems, stride
// multiple of 8 so ds_load_b128 stays 16B aligned). CDNA5 layout: lanes 0-15
// carry rows; half=lane>>4 selects K-halves (k0+half*8 .. / k0+16+half*8 ..).
__device__ __forceinline__ v16bf frag_lds(const unsigned short* p, int rowBase,
                                          int stride, int k0) {
  const int lane = threadIdx.x & 31;
  const int hs = lane >> 4;
  const unsigned short* q = p + (rowBase + (lane & 15)) * stride + k0 + hs * 8;
  FragU f;
  f.q[0] = *(const uint4*)(q);
  f.q[1] = *(const uint4*)(q + 16);
  return f.v;
}

__device__ __forceinline__ v16bf frag_glb(const unsigned short* base,
                                          long rowStride, int k0) {
  const int lane = threadIdx.x & 31;
  const int hs = lane >> 4;
  const unsigned short* q = base + (long)(lane & 15) * rowStride + k0 + hs * 8;
  FragU f;
  f.q[0] = *(const uint4*)(q);
  f.q[1] = *(const uint4*)(q + 16);
  return f.v;
}

// ---------------------------------------------------------------------------
// out[M,N] = X[M,K] @ W[N,K]^T, optional fused per-128-col RMSNorm.
// Tiles 128x128x64, 256 threads = 8 waves (2Mx4N), wave computes 64x32.
// ---------------------------------------------------------------------------
template <bool XBF16, bool NORM, bool OUTF32>
__global__ __launch_bounds__(256) void gemm_bt_kernel(
    const void* __restrict__ Xv, const float* __restrict__ W,
    const float* __restrict__ nw, void* __restrict__ outv,
    int M, int N, int K) {
  __shared__ unsigned short ldsA[128 * 72];
  __shared__ unsigned short ldsB[128 * 72];
  __shared__ float rowssq[128];
  __shared__ float rinv[128];

  const int tid  = threadIdx.x;
  const int wave = tid >> 5;
  const int wm   = wave >> 2;
  const int wn   = wave & 3;
  const int lane = tid & 31;
  const int blockM = blockIdx.y * 128;
  const int blockN = blockIdx.x * 128;

  const float* Xf = (const float*)Xv;
  const unsigned short* Xb = (const unsigned short*)Xv;

  v8f acc[4][2];
#pragma unroll
  for (int i = 0; i < 4; ++i)
#pragma unroll
    for (int j = 0; j < 2; ++j) acc[i][j] = (v8f){0.f,0.f,0.f,0.f,0.f,0.f,0.f,0.f};

  for (int k0 = 0; k0 < K; k0 += 64) {
    // ---- stage A tile (128 x 64) as bf16 ----
    if (XBF16) {
#if HAS_TDM
      if (wave == 0) {
        // 128 rows x 64 bf16; pad 4 DW (code 3) per 32 DW (code 4) -> stride 72
        tdm_load_2d(Xb + (long)blockM * K + k0, lds_off(ldsA),
                    64, 128, (unsigned)K, (unsigned)M, (unsigned)K, 4, 3);
        asm volatile("s_wait_tensorcnt 0x0" ::: "memory");
      }
#elif HAS_ASYNC
#pragma unroll
      for (int i = 0; i < 4; ++i) {
        int idx = tid + i * 256;                 // 1024 chunks of 8 bf16
        int row = idx >> 3, c8 = idx & 7;
        async_cp16(Xb + (long)(blockM + row) * K + k0 + c8 * 8,
                   ldsA + row * 72 + c8 * 8);
      }
      asm volatile("s_wait_asynccnt 0x0" ::: "memory");
#else
#pragma unroll
      for (int i = 0; i < 4; ++i) {
        int idx = tid + i * 256;
        int row = idx >> 3, c8 = idx & 7;
        uint4 d = *(const uint4*)(Xb + (long)(blockM + row) * K + k0 + c8 * 8);
        *(uint4*)(ldsA + row * 72 + c8 * 8) = d;
      }
#endif
    } else {
#pragma unroll
      for (int i = 0; i < 8; ++i) {
        int idx = tid + i * 256;                 // 2048 chunks of 4 f32
        int row = idx >> 4, c4 = idx & 15;
        float4 d = *(const float4*)(Xf + (long)(blockM + row) * K + k0 + c4 * 4);
        unsigned lo = (unsigned)f2bf(d.x) | ((unsigned)f2bf(d.y) << 16);
        unsigned hi = (unsigned)f2bf(d.z) | ((unsigned)f2bf(d.w) << 16);
        *(uint2*)(ldsA + row * 72 + c4 * 4) = make_uint2(lo, hi);
      }
    }
    // ---- stage B tile (128 N-rows x 64) from W (f32 row-major [N,K]) ----
#pragma unroll
    for (int i = 0; i < 8; ++i) {
      int idx = tid + i * 256;
      int row = idx >> 4, c4 = idx & 15;
      float4 d = *(const float4*)(W + (long)(blockN + row) * K + k0 + c4 * 4);
      unsigned lo = (unsigned)f2bf(d.x) | ((unsigned)f2bf(d.y) << 16);
      unsigned hi = (unsigned)f2bf(d.z) | ((unsigned)f2bf(d.w) << 16);
      *(uint2*)(ldsB + row * 72 + c4 * 4) = make_uint2(lo, hi);
    }
    // prefetch next K-tile of the streamed operands
    if (k0 + 64 < K) {
      int prow = tid >> 1, ph = tid & 1;
      __builtin_prefetch(W + (long)(blockN + prow) * K + k0 + 64 + ph * 32, 0, 1);
      if (!XBF16)
        __builtin_prefetch(Xf + (long)(blockM + prow) * K + k0 + 64 + ph * 32, 0, 1);
    }
    __syncthreads();

#pragma unroll
    for (int kk = 0; kk < 64; kk += 32) {
      v16bf afr[4], bfr[2];
#pragma unroll
      for (int mi = 0; mi < 4; ++mi)
        afr[mi] = frag_lds(ldsA, wm * 64 + mi * 16, 72, kk);
#pragma unroll
      for (int ni = 0; ni < 2; ++ni)
        bfr[ni] = frag_lds(ldsB, wn * 32 + ni * 16, 72, kk);
#pragma unroll
      for (int mi = 0; mi < 4; ++mi)
#pragma unroll
        for (int ni = 0; ni < 2; ++ni)
          acc[mi][ni] = __builtin_amdgcn_wmma_f32_16x16x32_bf16(
              false, afr[mi], false, bfr[ni], (short)0, acc[mi][ni], false, false);
    }
    __syncthreads();
  }

  if (NORM) {
    if (tid < 128) rowssq[tid] = 0.f;
    __syncthreads();
#pragma unroll
    for (int mi = 0; mi < 4; ++mi)
#pragma unroll
      for (int ni = 0; ni < 2; ++ni)
#pragma unroll
        for (int r = 0; r < 8; ++r) {
          int m = wm * 64 + mi * 16 + r + 8 * (lane >> 4);
          float v = acc[mi][ni][r];
          atomicAdd(&rowssq[m], v * v);          // ds_add_f32
        }
    __syncthreads();
    if (tid < 128) rinv[tid] = rsqrtf(rowssq[tid] * (1.0f / 128.f) + EPSF);
    __syncthreads();
  }

#pragma unroll
  for (int mi = 0; mi < 4; ++mi)
#pragma unroll
    for (int ni = 0; ni < 2; ++ni)
#pragma unroll
      for (int r = 0; r < 8; ++r) {
        int m = wm * 64 + mi * 16 + r + 8 * (lane >> 4);
        int n = wn * 32 + ni * 16 + (lane & 15);
        float v = acc[mi][ni][r];
        if (NORM) v = v * rinv[m] * nw[n];
        long o = (long)(blockM + m) * N + blockN + n;
        if (OUTF32) __builtin_nontemporal_store(v, &((float*)outv)[o]);
        else        ((unsigned short*)outv)[o] = f2bf(v);
      }
}

// ---------------------------------------------------------------------------
// Flash attention: grid (S/128, H, B); 8 waves, each wave owns 16 queries.
// K tile staged via TDM (or async/plain fallback); V transposed manually.
// ---------------------------------------------------------------------------
__global__ __launch_bounds__(256) void attn_kernel(
    const unsigned short* __restrict__ Qn, const unsigned short* __restrict__ Kn,
    const unsigned short* __restrict__ Vn, unsigned short* __restrict__ Out) {
  __shared__ unsigned short ldsK[64 * 136];      // [key][dim]
  __shared__ unsigned short ldsVt[128 * 72];     // [dim][key] (transposed)
  __shared__ unsigned short ldsP[8 * 16 * 72];   // per-wave P tiles

  const int tid  = threadIdx.x;
  const int wave = tid >> 5;
  const int lane = tid & 31;
  const int h  = blockIdx.y;                     // h = kv*G + g
  const int kv = h >> 2;
  const int b  = blockIdx.z;
  const int qbase = blockIdx.x * 128 + wave * 16;
  const float scale = 0.08838834764831845f;      // 1/sqrt(128)

  v16bf qf[4];
  {
    const unsigned short* qptr = Qn + ((long)(b * SEQN + qbase)) * DIMN + h * HDN;
#pragma unroll
    for (int kc = 0; kc < 4; ++kc) qf[kc] = frag_glb(qptr, DIMN, kc * 32);
  }

  v8f accO[8];
#pragma unroll
  for (int i = 0; i < 8; ++i) accO[i] = (v8f){0.f,0.f,0.f,0.f,0.f,0.f,0.f,0.f};
  float rowmax[8], rowsum[8];
#pragma unroll
  for (int r = 0; r < 8; ++r) { rowmax[r] = -1e30f; rowsum[r] = 0.f; }

  const long kvOff = (long)b * SEQN * KVDN + kv * HDN;

  for (int kt = 0; kt < SEQN / 64; ++kt) {
    // ---- stage K tile [64 keys][128 dims] ----
    const unsigned short* kGlb = Kn + kvOff + (long)(kt * 64) * KVDN;
#if HAS_TDM
    if (wave == 0) {
      // 64 rows x 128 bf16; pad 4 DW (code 3) per 64 DW (code 5) -> stride 136
      tdm_load_2d(kGlb, lds_off(ldsK), 128, 64,
                  (unsigned)KVDN, (unsigned)SEQN, (unsigned)KVDN, 5, 3);
      asm volatile("s_wait_tensorcnt 0x0" ::: "memory");
    }
#elif HAS_ASYNC
#pragma unroll
    for (int i = 0; i < 4; ++i) {
      int idx = tid + i * 256;
      int row = idx >> 4, c8 = idx & 15;
      async_cp16(kGlb + (long)row * KVDN + c8 * 8, ldsK + row * 136 + c8 * 8);
    }
    asm volatile("s_wait_asynccnt 0x0" ::: "memory");
#else
#pragma unroll
    for (int i = 0; i < 4; ++i) {
      int idx = tid + i * 256;
      int row = idx >> 4, c8 = idx & 15;
      uint4 d = *(const uint4*)(kGlb + (long)row * KVDN + c8 * 8);
      *(uint4*)(ldsK + row * 136 + c8 * 8) = d;
    }
#endif
    // ---- stage V tile transposed -> [dim][key] ----
#pragma unroll
    for (int i = 0; i < 4; ++i) {
      int idx = tid + i * 256;
      int row = idx >> 4, c8 = idx & 15;
      uint4 d = *(const uint4*)(Vn + kvOff + (long)(kt * 64 + row) * KVDN + c8 * 8);
      union { uint4 q; unsigned short s[8]; } u; u.q = d;
#pragma unroll
      for (int j = 0; j < 8; ++j) ldsVt[(c8 * 8 + j) * 72 + row] = u.s[j];
    }
    __syncthreads();

    // ---- S = Q @ K^T (16 x 64) ----
    v8f accS[4];
#pragma unroll
    for (int ni = 0; ni < 4; ++ni) accS[ni] = (v8f){0.f,0.f,0.f,0.f,0.f,0.f,0.f,0.f};
#pragma unroll
    for (int kc = 0; kc < 4; ++kc)
#pragma unroll
      for (int ni = 0; ni < 4; ++ni) {
        v16bf kf = frag_lds(ldsK, ni * 16, 136, kc * 32);
        accS[ni] = __builtin_amdgcn_wmma_f32_16x16x32_bf16(
            false, qf[kc], false, kf, (short)0, accS[ni], false, false);
      }

    // ---- online softmax ----
    unsigned short* pbase = ldsP + wave * 16 * 72;
#pragma unroll
    for (int r = 0; r < 8; ++r) {
      float tmax = -1e30f;
#pragma unroll
      for (int ni = 0; ni < 4; ++ni) tmax = fmaxf(tmax, accS[ni][r] * scale);
#pragma unroll
      for (int off = 1; off < 16; off <<= 1)
        tmax = fmaxf(tmax, __shfl_xor(tmax, off, 32));
      float nmax  = fmaxf(rowmax[r], tmax);
      float alpha = __expf(rowmax[r] - nmax);
      rowmax[r] = nmax;
      int mrow = r + 8 * (lane >> 4);
      float psum = 0.f;
#pragma unroll
      for (int ni = 0; ni < 4; ++ni) {
        float p = __expf(accS[ni][r] * scale - nmax);
        psum += p;
        pbase[mrow * 72 + ni * 16 + (lane & 15)] = f2bf(p);
      }
#pragma unroll
      for (int off = 1; off < 16; off <<= 1) psum += __shfl_xor(psum, off, 32);
      rowsum[r] = rowsum[r] * alpha + psum;
#pragma unroll
      for (int no = 0; no < 8; ++no) accO[no][r] *= alpha;
    }
    __syncthreads();

    // ---- O += P @ V ----
    v16bf pf[2];
    pf[0] = frag_lds(pbase, 0, 72, 0);
    pf[1] = frag_lds(pbase, 0, 72, 32);
#pragma unroll
    for (int no = 0; no < 8; ++no)
#pragma unroll
      for (int kc = 0; kc < 2; ++kc) {
        v16bf vf = frag_lds(ldsVt, no * 16, 72, kc * 32);
        accO[no] = __builtin_amdgcn_wmma_f32_16x16x32_bf16(
            false, pf[kc], false, vf, (short)0, accO[no], false, false);
      }
    __syncthreads();
  }

#pragma unroll
  for (int no = 0; no < 8; ++no)
#pragma unroll
    for (int r = 0; r < 8; ++r) {
      int mrow = r + 8 * (lane >> 4);
      float v = accO[no][r] / rowsum[r];
      long qrow = (long)(b * SEQN + qbase + mrow);
      Out[qrow * DIMN + h * HDN + no * 16 + (lane & 15)] = f2bf(v);
    }
}

// ---------------------------------------------------------------------------
extern "C" void kernel_launch(void* const* d_in, const int* in_sizes, int n_in,
                              void* d_out, int out_size, void* d_ws, size_t ws_size,
                              hipStream_t stream) {
  const float* query = (const float*)d_in[0];
  const float* key   = (const float*)d_in[1];
  const float* value = (const float*)d_in[2];
  const float* Wq    = (const float*)d_in[3];
  const float* Wk    = (const float*)d_in[4];
  const float* Wv    = (const float*)d_in[5];
  const float* Wo    = (const float*)d_in[6];
  const float* nqw   = (const float*)d_in[7];
  const float* nkw   = (const float*)d_in[8];

  const int M = BATCHN * SEQN;                  // 8192
  unsigned short* Qn = (unsigned short*)d_ws;
  unsigned short* Kn = Qn + (size_t)M * DIMN;
  unsigned short* Vn = Kn + (size_t)M * KVDN;
  unsigned short* At = Vn + (size_t)M * KVDN;

  dim3 blk(256);
  gemm_bt_kernel<false, true, false><<<dim3(DIMN / 128, M / 128), blk, 0, stream>>>(
      query, Wq, nqw, Qn, M, DIMN, DIMN);
  gemm_bt_kernel<false, true, false><<<dim3(KVDN / 128, M / 128), blk, 0, stream>>>(
      key, Wk, nkw, Kn, M, KVDN, DIMN);
  gemm_bt_kernel<false, false, false><<<dim3(KVDN / 128, M / 128), blk, 0, stream>>>(
      value, Wv, nullptr, Vn, M, KVDN, DIMN);
  attn_kernel<<<dim3(SEQN / 128, HEADS, BATCHN), blk, 0, stream>>>(Qn, Kn, Vn, At);
  gemm_bt_kernel<true, false, true><<<dim3(DIMN / 128, M / 128), blk, 0, stream>>>(
      At, Wo, nullptr, d_out, M, DIMN, DIMN);
}
